// _LinOSSBlock_89928025244482
// MI455X (gfx1250) — matrix operand order repaired and checked by reading, not compiled, not run on hardware
//
#include <hip/hip_runtime.h>
#include <hip/hip_bf16.h>

// LinOSS block for MI455X (gfx1250, wave32, WMMA).
// Memory-bound (~1.0 GB traffic => ~45us floor @ 23.3 TB/s). GEMMs use
// v_wmma_f32_16x16x32_f16 (f32 accum). Weights pre-converted to f16 once;
// per-block staging is b128 global->LDS copy, double-buffered (1 barrier/k-step).

typedef __attribute__((ext_vector_type(16))) _Float16 v16h;
typedef __attribute__((ext_vector_type(8)))  _Float16 v8h;
typedef __attribute__((ext_vector_type(4)))  _Float16 v4h;
typedef __attribute__((ext_vector_type(8)))  float    v8f;

#define Bsz   32
#define Tn    4096
#define Hn    256
#define Sn    256
#define Rn    (Bsz * Tn)        // 131072 rows
#define CHUNK 64
#define NCH   (Tn / CHUNK)      // 64 chunks
#define LDH   264               // padded half stride for row-major A tiles
#define LDW   40                // padded half stride for transposed B slices
#define WTSZ  (256 * LDW)       // halfs per weight-slice buffer

// ---- WMMA fragment loaders (layouts per CDNA5 ISA 7.12.2, 16-bit, wave32) ---

// A: 16x32 f16 from row-major LDS. lane L: row=L&15, half=L>>4.
// a[0..7] = K 8h..8h+7 ; a[8..15] = K 16+8h..+7  -> two 16B LDS loads.
__device__ __forceinline__ v16h load_a_frag(const _Float16* sm, int row0, int k0) {
  const int lane = threadIdx.x & 31;
  const int r = lane & 15, hh = lane >> 4;
  const _Float16* base = sm + (row0 + r) * LDH + k0 + 8 * hh;
  v8h lo = *(const v8h*)(base);
  v8h hi = *(const v8h*)(base + 16);
  return __builtin_shufflevector(lo, hi, 0,1,2,3,4,5,6,7,8,9,10,11,12,13,14,15);
}

// B: 32x16 f16 from LDS staged transposed as s_wt[n][k] (stride LDW).
// lane L: col=L&15, half=L>>4; b[0..7] = K 8h..8h+7 ; b[8..15] = K 16+8h..+7 -> two 16B LDS loads.
__device__ __forceinline__ v16h load_b_frag(const _Float16* swt, int n0) {
  const int lane = threadIdx.x & 31;
  const int n = lane & 15, hh = lane >> 4;
  const _Float16* base = swt + (n0 + n) * LDW + 8 * hh;
  v8h lo = *(const v8h*)(base);
  v8h hi = *(const v8h*)(base + 16);
  return __builtin_shufflevector(lo, hi, 0,1,2,3,4,5,6,7,8,9,10,11,12,13,14,15);
}

// Stage a 32(K) x 256(N) f16 weight slice into LDS (pure b128 copy).
// B[k][n] = W16[n*ld + k0 + k]; thread n=tid copies 32 contiguous halves.
__device__ __forceinline__ void stage_wt16(_Float16* s_wt,
                                           const _Float16* __restrict__ w16,
                                           int ld, int k0) {
  const int tid = threadIdx.x;
  const v8h* src = (const v8h*)(w16 + (size_t)tid * ld + k0);
  v8h* dst = (v8h*)(s_wt + tid * LDW);
#pragma unroll
  for (int i = 0; i < 4; ++i) dst[i] = src[i];
}

// ---- Kernel 0a: recurrence coefficients ------------------------------------

__device__ __forceinline__ float softplusf(float x) {
  return (x > 20.f) ? x : log1pf(expf(x));
}

__global__ __launch_bounds__(256) void k_coeff(const float* __restrict__ a_diag,
                                               const float* __restrict__ g_diag,
                                               const float* __restrict__ dt,
                                               float* __restrict__ coeff,
                                               float* __restrict__ coeffL) {
  int s = threadIdx.x;
  float dt_s  = softplusf(dt[s]) + 1e-4f;
  float omega = a_diag[s] * dt_s;
  float dec   = expf(-softplusf(g_diag[s]) * dt_s);
  dec *= dec;                              // damped=True
  float c = dec * cosf(omega);
  coeff[s] = c;
  float p = c;                             // coeff^CHUNK, CHUNK = 64 = 2^6
#pragma unroll
  for (int i = 0; i < 6; ++i) p *= p;
  coeffL[s] = p;
}

// ---- Kernel 0b: one-time weight f32 -> f16 conversion ----------------------

__global__ __launch_bounds__(256) void k_cvt_w(const float4* __restrict__ a,
                                               const float4* __restrict__ b,
                                               const float4* __restrict__ c,
                                               v4h* __restrict__ a16,
                                               v4h* __restrict__ b16,
                                               v4h* __restrict__ c16) {
  int i = blockIdx.x * 256 + threadIdx.x;   // 0..16383 (Hn*Sn/4)
  float4 v; v4h h;
  v = a[i]; h[0]=(_Float16)v.x; h[1]=(_Float16)v.y; h[2]=(_Float16)v.z; h[3]=(_Float16)v.w; a16[i]=h;
  v = b[i]; h[0]=(_Float16)v.x; h[1]=(_Float16)v.y; h[2]=(_Float16)v.z; h[3]=(_Float16)v.w; b16[i]=h;
  v = c[i]; h[0]=(_Float16)v.x; h[1]=(_Float16)v.y; h[2]=(_Float16)v.z; h[3]=(_Float16)v.w; c16[i]=h;
}

// ---- Kernel 1: fused LayerNorm + u = xn @ in_to_state^T --------------------

__global__ __launch_bounds__(256) void k_ln_gemm1(const float* __restrict__ x,
                                                  const _Float16* __restrict__ w16,
                                                  const float* __restrict__ nw,
                                                  const float* __restrict__ nb,
                                                  _Float16* __restrict__ xn_g,
                                                  float* __restrict__ u) {
  __shared__ float    s_x[32 * Hn];        // x tile, reused for u epilogue
  __shared__ _Float16 s_xn[32 * LDH];
  __shared__ _Float16 s_wt[2 * WTSZ];      // double-buffered weight slices

  const int tid  = threadIdx.x;
  const int lane = tid & 31;
  const int wave = tid >> 5;
  const size_t r0 = (size_t)blockIdx.x * 32;

  // Load 32x256 x tile as float4 (b128)
  {
    const float4* xp = (const float4*)(x + r0 * Hn);
    float4* sx4 = (float4*)s_x;
#pragma unroll
    for (int i = 0; i < 8; ++i) sx4[tid + i * 256] = xp[tid + i * 256];
  }
  __syncthreads();

  // LayerNorm: wave w normalizes rows 4w..4w+3 via shuffle reduction
#pragma unroll
  for (int rr = 0; rr < 4; ++rr) {
    int r = wave * 4 + rr;
    float sum = 0.f, sq = 0.f;
#pragma unroll
    for (int j = 0; j < 8; ++j) {
      float v = s_x[r * Hn + lane + 32 * j];
      sum += v; sq += v * v;
    }
#pragma unroll
    for (int off = 16; off > 0; off >>= 1) {
      sum += __shfl_xor(sum, off, 32);
      sq  += __shfl_xor(sq,  off, 32);
    }
    float mean = sum * (1.f / Hn);
    float var  = sq * (1.f / Hn) - mean * mean;
    float rstd = rsqrtf(var + 1e-5f);
#pragma unroll
    for (int j = 0; j < 8; ++j) {
      int c = lane + 32 * j;
      float v = (s_x[r * Hn + c] - mean) * rstd * nw[c] + nb[c];
      s_xn[r * LDH + c] = (_Float16)v;
    }
  }
  __syncthreads();

  // Copy xn tile to global as 16B vectors (thread -> 32 contiguous halves)
  {
    int r = tid >> 3, c0 = (tid & 7) * 32;
    const v8h* src = (const v8h*)(s_xn + r * LDH + c0);
    v8h* dst = (v8h*)(xn_g + (r0 + r) * Hn + c0);
#pragma unroll
    for (int i = 0; i < 4; ++i) dst[i] = src[i];
  }

  // GEMM: 32x256 output; wave w owns row strip (w>>2) x 4 col tiles
  const int rs = wave >> 2;
  const int cg = wave & 3;
  v8f acc[4];
#pragma unroll
  for (int j = 0; j < 4; ++j)
#pragma unroll
    for (int v = 0; v < 8; ++v) acc[j][v] = 0.f;

  stage_wt16(s_wt, w16, Hn, 0);
  __syncthreads();
  for (int k0 = 0; k0 < Hn; k0 += 32) {
    const int cur = (k0 >> 5) & 1;
    if (k0 + 32 < Hn) stage_wt16(s_wt + (cur ^ 1) * WTSZ, w16, Hn, k0 + 32);
    v16h a = load_a_frag(s_xn, rs * 16, k0);
    const _Float16* wb = s_wt + cur * WTSZ;
#pragma unroll
    for (int j = 0; j < 4; ++j) {
      v16h b = load_b_frag(wb, (cg * 4 + j) * 16);
      acc[j] = __builtin_amdgcn_wmma_f32_16x16x32_f16(false, a, false, b,
                                                      (short)0, acc[j], false, false);
    }
    __syncthreads();
  }

  // Epilogue via LDS (reuse s_x) -> float4 global stores
  const int hh = lane >> 4, nn = lane & 15;
#pragma unroll
  for (int j = 0; j < 4; ++j) {
    int col = (cg * 4 + j) * 16 + nn;
#pragma unroll
    for (int v = 0; v < 8; ++v) {
      int lr = rs * 16 + v + 8 * hh;
      s_x[lr * Sn + col] = acc[j][v];
    }
  }
  __syncthreads();
  {
    const float4* sx4 = (const float4*)s_x;
    float4* up = (float4*)(u + r0 * Sn);
#pragma unroll
    for (int i = 0; i < 8; ++i) up[tid + i * 256] = sx4[tid + i * 256];
  }
}

// ---- Kernels 2a/2b/2c: chunked diagonal scan over T (float4 over S) --------

__global__ __launch_bounds__(64) void k_scan_local(const float4* __restrict__ coeff4,
                                                   float4* __restrict__ u4,
                                                   float4* __restrict__ carries4) {
  const int s4 = threadIdx.x;            // 0..63 (4 channels each)
  const int c = blockIdx.x, b = blockIdx.y;
  const float4 cf = coeff4[s4];
  size_t base = ((size_t)b * Tn + (size_t)c * CHUNK) * (Sn / 4) + s4;
  float4 val = make_float4(0.f, 0.f, 0.f, 0.f);
#pragma unroll 4
  for (int t = 0; t < CHUNK; ++t) {
    float4 uu = u4[base];
    val.x = fmaf(cf.x, val.x, uu.x);
    val.y = fmaf(cf.y, val.y, uu.y);
    val.z = fmaf(cf.z, val.z, uu.z);
    val.w = fmaf(cf.w, val.w, uu.w);
    u4[base] = val;                      // in-place local scan
    base += Sn / 4;
  }
  carries4[((size_t)b * NCH + c) * (Sn / 4) + s4] = val;
}

__global__ __launch_bounds__(64) void k_scan_carry(const float4* __restrict__ coeffL4,
                                                   const float4* __restrict__ state0_4,
                                                   float4* __restrict__ carries4,
                                                   float4* __restrict__ final4) {
  const int s4 = threadIdx.x;
  const int b = blockIdx.x;
  const float4 cl = coeffL4[s4];
  float4 st = state0_4[b * (Sn / 4) + s4];
  for (int c = 0; c < NCH; ++c) {
    size_t idx = ((size_t)b * NCH + c) * (Sn / 4) + s4;
    float4 raw = carries4[idx];
    carries4[idx] = st;                  // carry-in for chunk c
    st.x = fmaf(cl.x, st.x, raw.x);
    st.y = fmaf(cl.y, st.y, raw.y);
    st.z = fmaf(cl.z, st.z, raw.z);
    st.w = fmaf(cl.w, st.w, raw.w);
  }
  final4[b * (Sn / 4) + s4] = st;        // second tuple output
}

__global__ __launch_bounds__(64) void k_scan_apply(const float4* __restrict__ coeff4,
                                                   float4* __restrict__ u4,
                                                   const float4* __restrict__ carries4) {
  const int s4 = threadIdx.x;
  const int c = blockIdx.x, b = blockIdx.y;
  const float4 cf = coeff4[s4];
  const float4 cin = carries4[((size_t)b * NCH + c) * (Sn / 4) + s4];
  size_t base = ((size_t)b * Tn + (size_t)c * CHUNK) * (Sn / 4) + s4;
  float4 pw = cf;
#pragma unroll 4
  for (int t = 0; t < CHUNK; ++t) {
    float4 uu = u4[base];
    uu.x = fmaf(pw.x, cin.x, uu.x);
    uu.y = fmaf(pw.y, cin.y, uu.y);
    uu.z = fmaf(pw.z, cin.z, uu.z);
    uu.w = fmaf(pw.w, cin.w, uu.w);
    u4[base] = uu;                       // s_t = local_t + coeff^{t+1} * carry
    pw.x *= cf.x; pw.y *= cf.y; pw.z *= cf.z; pw.w *= cf.w;
    base += Sn / 4;
  }
}

// ---- Kernel 3: fused readout GEMM + gelu + out GEMM + residual -------------

__global__ __launch_bounds__(256) void k_readout(const float* __restrict__ st,
                                                 const _Float16* __restrict__ xn_g,
                                                 const _Float16* __restrict__ sth16,
                                                 const float* __restrict__ direct,
                                                 const _Float16* __restrict__ ow16,
                                                 const float* __restrict__ ob,
                                                 const float* __restrict__ x,
                                                 float* __restrict__ y) {
  __shared__ _Float16 s_m[32 * LDH];     // states, then reused for mixed
  __shared__ _Float16 s_wt[2 * WTSZ];    // double-buffered weight slices
  __shared__ float    s_out[32 * Hn];    // y staging for float4 stores

  const int tid  = threadIdx.x;
  const int lane = tid & 31;
  const int wave = tid >> 5;
  const size_t r0 = (size_t)blockIdx.x * 32;

  // Stage states tile: float4 global loads -> f16 LDS (8B stores)
  {
    const float4* sp = (const float4*)(st + r0 * Sn);
#pragma unroll
    for (int i = 0; i < 8; ++i) {
      int idx = tid + i * 256;
      float4 v = sp[idx];
      int r = idx >> 6, c = (idx & 63) * 4;
      v4h h; h[0] = (_Float16)v.x; h[1] = (_Float16)v.y;
             h[2] = (_Float16)v.z; h[3] = (_Float16)v.w;
      *(v4h*)(s_m + r * LDH + c) = h;
    }
  }

  const int rs = wave >> 2;
  const int cg = wave & 3;
  const int hh = lane >> 4, nn = lane & 15;

  // GEMM1: mixed = states @ state_to_hidden^T
  v8f acc[4];
#pragma unroll
  for (int j = 0; j < 4; ++j)
#pragma unroll
    for (int v = 0; v < 8; ++v) acc[j][v] = 0.f;

  stage_wt16(s_wt, sth16, Sn, 0);
  __syncthreads();
  for (int k0 = 0; k0 < Sn; k0 += 32) {
    const int cur = (k0 >> 5) & 1;
    if (k0 + 32 < Sn) stage_wt16(s_wt + (cur ^ 1) * WTSZ, sth16, Sn, k0 + 32);
    v16h a = load_a_frag(s_m, rs * 16, k0);
    const _Float16* wb = s_wt + cur * WTSZ;
#pragma unroll
    for (int j = 0; j < 4; ++j) {
      v16h b = load_b_frag(wb, (cg * 4 + j) * 16);
      acc[j] = __builtin_amdgcn_wmma_f32_16x16x32_f16(false, a, false, b,
                                                      (short)0, acc[j], false, false);
    }
    __syncthreads();
  }

  // Epilogue 1: + direct * xn, tanh-gelu; write mixed (f16) back into s_m.
  // (All s_m reads finished at the loop's final barrier.)
#pragma unroll
  for (int j = 0; j < 4; ++j) {
    int col = (cg * 4 + j) * 16 + nn;
#pragma unroll
    for (int v = 0; v < 8; ++v) {
      int lr = rs * 16 + v + 8 * hh;
      float m = acc[j][v] + direct[col] * (float)xn_g[(r0 + lr) * Hn + col];
      float inner = 0.7978845608028654f * (m + 0.044715f * m * m * m);
      m = 0.5f * m * (1.f + tanhf(inner));
      s_m[lr * LDH + col] = (_Float16)m;
    }
  }

  // GEMM2: y = mixed @ out_w^T
  v8f acc2[4];
#pragma unroll
  for (int j = 0; j < 4; ++j)
#pragma unroll
    for (int v = 0; v < 8; ++v) acc2[j][v] = 0.f;

  stage_wt16(s_wt, ow16, Hn, 0);
  __syncthreads();                       // also publishes mixed tile
  for (int k0 = 0; k0 < Hn; k0 += 32) {
    const int cur = (k0 >> 5) & 1;
    if (k0 + 32 < Hn) stage_wt16(s_wt + (cur ^ 1) * WTSZ, ow16, Hn, k0 + 32);
    v16h a = load_a_frag(s_m, rs * 16, k0);
    const _Float16* wb = s_wt + cur * WTSZ;
#pragma unroll
    for (int j = 0; j < 4; ++j) {
      v16h b = load_b_frag(wb, (cg * 4 + j) * 16);
      acc2[j] = __builtin_amdgcn_wmma_f32_16x16x32_f16(false, a, false, b,
                                                       (short)0, acc2[j], false, false);
    }
    __syncthreads();
  }

  // Epilogue 2: (+ out_b) -> LDS; then float4 copy-out adding residual x
#pragma unroll
  for (int j = 0; j < 4; ++j) {
    int col = (cg * 4 + j) * 16 + nn;
#pragma unroll
    for (int v = 0; v < 8; ++v) {
      int lr = rs * 16 + v + 8 * hh;
      s_out[lr * Hn + col] = acc2[j][v] + ob[col];
    }
  }
  __syncthreads();
  {
    const float4* so4 = (const float4*)s_out;
    const float4* xp  = (const float4*)(x + r0 * Hn);
    float4* yp        = (float4*)(y + r0 * Hn);
#pragma unroll
    for (int i = 0; i < 8; ++i) {
      int idx = tid + i * 256;
      float4 a = so4[idx], bx = xp[idx];
      a.x += bx.x; a.y += bx.y; a.z += bx.z; a.w += bx.w;
      yp[idx] = a;
    }
  }
}

// ---- Host launcher ---------------------------------------------------------

extern "C" void kernel_launch(void* const* d_in, const int* in_sizes, int n_in,
                              void* d_out, int out_size, void* d_ws, size_t ws_size,
                              hipStream_t stream) {
  const float* x       = (const float*)d_in[0];
  const float* state0  = (const float*)d_in[1];
  const float* w_in    = (const float*)d_in[2];   // in_to_state [S,H]
  const float* sth     = (const float*)d_in[3];   // state_to_hidden [H,S]
  const float* direct  = (const float*)d_in[4];
  const float* a_diag  = (const float*)d_in[5];
  const float* g_diag  = (const float*)d_in[6];
  const float* dt      = (const float*)d_in[7];
  const float* norm_w  = (const float*)d_in[8];
  const float* norm_b  = (const float*)d_in[9];
  const float* out_w   = (const float*)d_in[10];  // [H,H]
  const float* out_b   = (const float*)d_in[11];

  float* y           = (float*)d_out;
  float* final_state = y + (size_t)Bsz * Tn * Hn;

  // Workspace (~204 MB): u/states f32 (scan in-place), xn f16, carries,
  // coeffs, f16 weight copies.
  char* ws = (char*)d_ws;
  float*    u       = (float*)ws;      ws += (size_t)Rn * Sn * sizeof(float);
  _Float16* xn      = (_Float16*)ws;   ws += (size_t)Rn * Hn * sizeof(_Float16);
  float*    carries = (float*)ws;      ws += (size_t)Bsz * NCH * Sn * sizeof(float);
  float*    coeff   = (float*)ws;      ws += Sn * sizeof(float);
  float*    coeffL  = (float*)ws;      ws += Sn * sizeof(float);
  _Float16* w16_in  = (_Float16*)ws;   ws += (size_t)Sn * Hn * sizeof(_Float16);
  _Float16* w16_sth = (_Float16*)ws;   ws += (size_t)Hn * Sn * sizeof(_Float16);
  _Float16* w16_ow  = (_Float16*)ws;   ws += (size_t)Hn * Hn * sizeof(_Float16);

  k_coeff <<<1, 256, 0, stream>>>(a_diag, g_diag, dt, coeff, coeffL);
  k_cvt_w <<<Hn * Sn / 4 / 256, 256, 0, stream>>>(
      (const float4*)w_in, (const float4*)sth, (const float4*)out_w,
      (v4h*)w16_in, (v4h*)w16_sth, (v4h*)w16_ow);
  k_ln_gemm1  <<<Rn / 32, 256, 0, stream>>>(x, w16_in, norm_w, norm_b, xn, u);
  k_scan_local<<<dim3(NCH, Bsz), 64, 0, stream>>>((const float4*)coeff,
                                                  (float4*)u, (float4*)carries);
  k_scan_carry<<<Bsz, 64, 0, stream>>>((const float4*)coeffL, (const float4*)state0,
                                       (float4*)carries, (float4*)final_state);
  k_scan_apply<<<dim3(NCH, Bsz), 64, 0, stream>>>((const float4*)coeff,
                                                  (float4*)u, (const float4*)carries);
  k_readout   <<<Rn / 32, 256, 0, stream>>>(u, xn, w16_sth, direct, w16_ow,
                                            out_b, x, y);
}
